// SuEIR_39453569581647
// MI455X (gfx1250) — matrix operand order repaired and checked by reading, not compiled, not run on hardware
//
#include <hip/hip_runtime.h>

typedef _Float16 f16;
typedef __attribute__((ext_vector_type(16))) _Float16 v16h;
typedef __attribute__((ext_vector_type(8)))  _Float16 f16x8;
typedef __attribute__((ext_vector_type(8)))  float    v8f;
typedef __attribute__((ext_vector_type(4)))  float    f32x4;

#define NR 4096   // regions
#define NT 1024   // steps

// ---------------------------------------------------------------------------
// Convert A (f32 row-major) -> f16 row-major. 8 elements / thread.
// ---------------------------------------------------------------------------
__global__ __launch_bounds__(256)
void cvtA(const float* __restrict__ A, f16* __restrict__ A16) {
    const size_t i = ((size_t)blockIdx.x * blockDim.x + threadIdx.x) * 8;
    f32x4 a0 = *(const f32x4*)(A + i);
    f32x4 a1 = *(const f32x4*)(A + i + 4);
    f16x8 o;
    o[0] = (f16)a0[0]; o[1] = (f16)a0[1]; o[2] = (f16)a0[2]; o[3] = (f16)a0[3];
    o[4] = (f16)a1[0]; o[5] = (f16)a1[1]; o[6] = (f16)a1[2]; o[7] = (f16)a1[3];
    *(f16x8*)(A16 + i) = o;
}

// ---------------------------------------------------------------------------
// Init: copy state to workspace, x0 = (I+E)*S in f16, write t=0 outputs.
// ---------------------------------------------------------------------------
__global__ __launch_bounds__(256)
void initk(const float* __restrict__ S0, const float* __restrict__ E0,
           const float* __restrict__ I0, const float* __restrict__ R0,
           float* __restrict__ S, float* __restrict__ E,
           float* __restrict__ I, float* __restrict__ Rr,
           f16* __restrict__ x0, float* __restrict__ out) {
    const int r = blockIdx.x * blockDim.x + threadIdx.x;
    float s = S0[r], e = E0[r], ii = I0[r], rc = R0[r];
    S[r] = s; E[r] = e; I[r] = ii; Rr[r] = rc;
    x0[r] = (f16)((ii + e) * s);
    out[(size_t)r * (2 * NT) + 0] = ii;
    out[(size_t)r * (2 * NT) + 1] = rc;
}

// ---------------------------------------------------------------------------
// One Euler step. 1 wave per block, 16 rows per wave, grid = NR/16 = 256.
// GEMV via v_wmma_f32_16x16x32_f16 with x broadcast to all 16 B-columns.
// ---------------------------------------------------------------------------
__global__ __launch_bounds__(32)
void stepk(const f16* __restrict__ A16,
           const f16* __restrict__ xin, f16* __restrict__ xout,
           float* __restrict__ S, float* __restrict__ E,
           float* __restrict__ I, float* __restrict__ Rr,
           const float* __restrict__ beta, const float* __restrict__ gamma,
           const float* __restrict__ sigma, const float* __restrict__ mu,
           const float* __restrict__ hstep, float* __restrict__ out, int t) {
    const int lane    = threadIdx.x;        // 0..31 (wave32)
    const int m       = lane & 15;          // matrix row within tile
    const int khi     = lane >> 4;          // K-half select (0 or 1)
    const int rowBase = blockIdx.x * 16;

    // 16-bit A-matrix 16x32 layout: lanes 0-15 = M, hold K=0..7 (v0-3) and
    // K=16..23 (v4-7); lanes 16-31 = same M, K offset +8.
    const f16* __restrict__ arow = A16 + (size_t)(rowBase + m) * NR + khi * 8;
    // 16-bit B-matrix 32x16 layout: same K order as A (lanes 0-15 K=0..7,16..23;
    // lanes 16-31 K offset +8); N = lane&15 (we replicate x across all N columns).
    const f16* __restrict__ bptr = xin + khi * 8;

    v8f c = {};
    #pragma unroll 4
    for (int k = 0; k < NR; k += 32) {
        union { v16h v; struct { f16x8 lo, hi; } s; } a, b;
        a.s.lo = *(const f16x8*)(arow + k);        // K = k + khi*8 + [0..7]
        a.s.hi = *(const f16x8*)(arow + k + 16);   // K = k + 16 + khi*8 + [0..7]
        b.s.lo = *(const f16x8*)(bptr + k);        // K = k + khi*8 + [0..7]
        b.s.hi = *(const f16x8*)(bptr + k + 16);   // K = k + 16 + khi*8 + [0..7]
        c = __builtin_amdgcn_wmma_f32_16x16x32_f16(
                false, a.v, false, b.v, (short)0, c, false, false);
    }

    // D layout: VGPR j, lanes 0-15 -> M=j; lanes 16-31 -> M=j+8. All N columns
    // identical, so lane 0 updates rows 0..7, lane 16 updates rows 8..15.
    if (m == 0) {
        const float h = hstep[0];
        const int ro = rowBase + khi * 8;
        #pragma unroll
        for (int j = 0; j < 8; ++j) {
            const int r  = ro + j;
            const float av = c[j];
            const float s  = S[r],    e  = E[r];
            const float ii = I[r],    rc = Rr[r];
            const float be = beta[r], ga = gamma[r];
            const float si = sigma[r], muv = mu[r];
            const float ie   = ii + e;
            const float s_n  = s  - be * av * h;
            const float e_n  = e  + (be * s * ie - si * e) * h;
            const float i_n  = ii + (muv * si * e - ga * ii) * h;
            const float r_n  = rc + ga * ii * h;
            S[r] = s_n; E[r] = e_n; I[r] = i_n; Rr[r] = r_n;
            xout[r] = (f16)((i_n + e_n) * s_n);
            const size_t o = (size_t)r * (2 * NT) + (size_t)t * 2;
            out[o]     = i_n;
            out[o + 1] = r_n;
        }
    }
}

// ---------------------------------------------------------------------------
// Inputs: 0=num_steps(int), 1=init_S, 2=init_E, 3=init_I, 4=init_R,
//         5=A, 6=beta, 7=gamma, 8=sigma, 9=mu, 10=step
// Output: f32 [4096, 1024, 2]
// ---------------------------------------------------------------------------
extern "C" void kernel_launch(void* const* d_in, const int* in_sizes, int n_in,
                              void* d_out, int out_size, void* d_ws, size_t ws_size,
                              hipStream_t stream) {
    const float* init_S = (const float*)d_in[1];
    const float* init_E = (const float*)d_in[2];
    const float* init_I = (const float*)d_in[3];
    const float* init_R = (const float*)d_in[4];
    const float* A      = (const float*)d_in[5];
    const float* beta   = (const float*)d_in[6];
    const float* gamma  = (const float*)d_in[7];
    const float* sigma  = (const float*)d_in[8];
    const float* mu     = (const float*)d_in[9];
    const float* hstep  = (const float*)d_in[10];
    float* out = (float*)d_out;

    // Workspace layout
    char* ws = (char*)d_ws;
    f16*   A16 = (f16*)ws;                                   // 32 MB
    f16*   x0  = (f16*)(ws + (size_t)NR * NR * sizeof(f16)); // 8 KB
    f16*   x1  = x0 + NR;                                    // 8 KB
    float* S   = (float*)(x1 + NR);
    float* E   = S + NR;
    float* I   = E + NR;
    float* Rr  = I + NR;

    // A f32 -> f16 (re-done every launch; deterministic, ~4 us of HBM traffic)
    cvtA<<<(NR * (size_t)NR) / (256 * 8), 256, 0, stream>>>(A, A16);

    // State init + t=0 outputs + x0
    initk<<<NR / 256, 256, 0, stream>>>(init_S, init_E, init_I, init_R,
                                        S, E, I, Rr, x0, out);

    // 1023 Euler steps; launch boundary = grid-wide barrier; x double-buffered.
    for (int t = 1; t < NT; ++t) {
        f16* xin  = (t & 1) ? x0 : x1;
        f16* xout = (t & 1) ? x1 : x0;
        stepk<<<NR / 16, 32, 0, stream>>>(A16, xin, xout, S, E, I, Rr,
                                          beta, gamma, sigma, mu, hstep, out, t);
    }
}
